// GNN_drug_73959336837163
// MI455X (gfx1250) — matrix operand order repaired and checked by reading, not compiled, not run on hardware
//
#include <hip/hip_runtime.h>
#include <math.h>

// ---------------- problem constants ----------------
#define N_NODES  200000
#define N_EDGES  1600000
#define N_GRAPHS 4096
#define IN_DIM   75
#define HID      128
#define N_LAYERS 3
#define OUT_STRIDE (HID * N_LAYERS)   // 384
#define BN_EPS   1e-5f

// ---------------- CDNA5 WMMA types ----------------
typedef __attribute__((ext_vector_type(16))) __bf16          v16bf;
typedef __attribute__((ext_vector_type(8)))  float           v8f;
typedef __attribute__((ext_vector_type(8)))  unsigned short  v8us;

__device__ __forceinline__ unsigned short f32_bf16(float f) {
  unsigned int u = __float_as_uint(f);
  u = (u + 0x7FFFu + ((u >> 16) & 1u)) >> 16;   // round-to-nearest-even
  return (unsigned short)u;
}

// float atomic max via signed/unsigned integer punning (valid for any mix of signs)
__device__ __forceinline__ void atomicMaxF(float* addr, float val) {
  if (val >= 0.0f) atomicMax((int*)addr, __float_as_int(val));
  else             atomicMin((unsigned int*)addr, __float_as_uint(val));
}

// ---------------- small utility kernels ----------------
__global__ void fill_f32(float* __restrict__ p, long n, float v) {
  long i = (long)blockIdx.x * blockDim.x + threadIdx.x;
  if (i < n) p[i] = v;
}

// convert weight (rows x 128 f32, row-major K x N) -> TRANSPOSED padded bf16 (N x K, 128x128)
// out[n*128 + k] = bf16(w[k*128 + n]); zero pad for k >= rows
__global__ void cvt_w_bf16_t(const float* __restrict__ w, unsigned short* __restrict__ o, int rows) {
  int i = blockIdx.x * blockDim.x + threadIdx.x;
  if (i >= HID * HID) return;
  int n = i >> 7;            // output row   = N index
  int k = i & (HID - 1);     // output col   = K index
  float v = (k < rows) ? w[k * HID + n] : 0.0f;
  o[i] = f32_bf16(v);
}

// edge scatter: agg[dst] += x[src]. One block per edge, one thread per feature.
// adj indices are block-uniform -> scalar loads; feature access fully coalesced.
__global__ void scatter_add(const float* __restrict__ x, const int* __restrict__ adj,
                            float* __restrict__ agg, int d) {
  int e = blockIdx.x;
  int f = threadIdx.x;
  if (f >= d) return;
  int s = adj[e];
  int t = adj[N_EDGES + e];
  atomicAdd(&agg[(long)t * d + f], x[(long)s * d + f]);
}

// ---------------- fragment loaders (ISA §7.12.2 layouts), vectorized ----------------
// A tile is row-major [M][K] bf16, ld=128.
// A fragment (16x32): lane m (0-15): K in {k0+8h .. +7} U {k0+16+8h .. +7}  (h=half)
__device__ __forceinline__ v16bf load_a_frag(const unsigned short* __restrict__ a,
                                             int row0, int k0, int half, int m) {
  const unsigned short* rp = a + (row0 + m) * HID + k0 + 8 * half;
  union { v8us h[2]; v16bf v; } f;
  f.h[0] = *(const v8us*)(rp);        // 16B aligned
  f.h[1] = *(const v8us*)(rp + 16);   // 16B aligned
  return f.v;
}

// W tile is TRANSPOSED [N][K] bf16, ld=128.
// B fragment (32x16): lane n: K in {k0+8h .. +7} U {k0+16+8h .. +7}  (h=half)
__device__ __forceinline__ v16bf load_b_frag(const unsigned short* __restrict__ wt,
                                             int k0, int n0, int half, int n) {
  const unsigned short* cp = wt + (n0 + n) * HID + k0 + 8 * half;
  union { v8us h[2]; v16bf v; } f;
  f.h[0] = *(const v8us*)(cp);        // 16B aligned
  f.h[1] = *(const v8us*)(cp + 16);   // 16B aligned
  return f.v;
}

// full K=128 GEMM for one 16x128 wave tile, software double-buffered across k-steps
__device__ __forceinline__ void gemm128(const unsigned short* __restrict__ Alds,
                                        const unsigned short* __restrict__ Wlds,
                                        v8f acc[8], int row0, int half, int mn) {
  v16bf a = load_a_frag(Alds, row0, 0, half, mn);
  v16bf b[8];
#pragma unroll
  for (int nt = 0; nt < 8; ++nt) b[nt] = load_b_frag(Wlds, 0, nt * 16, half, mn);

#pragma unroll
  for (int k0 = 0; k0 < HID; k0 += 32) {
    v16bf a_n;
    v16bf b_n[8];
    if (k0 + 32 < HID) {                     // static after unroll
      a_n = load_a_frag(Alds, row0, k0 + 32, half, mn);
#pragma unroll
      for (int nt = 0; nt < 8; ++nt) b_n[nt] = load_b_frag(Wlds, k0 + 32, nt * 16, half, mn);
    }
#pragma unroll
    for (int nt = 0; nt < 8; ++nt) {
      acc[nt] = __builtin_amdgcn_wmma_f32_16x16x32_bf16(
          false, a, false, b[nt], (short)0, acc[nt], false, false);
    }
    if (k0 + 32 < HID) {
      a = a_n;
#pragma unroll
      for (int nt = 0; nt < 8; ++nt) b[nt] = b_n[nt];
    }
  }
}

// ---------------- fused MLP: h = relu(relu((x+agg)W1+b1)W2+b2) ----------------
__global__ __launch_bounds__(256, 1)
void gemm_mlp(const float* __restrict__ x, const float* __restrict__ agg,
              const unsigned short* __restrict__ w1t, const unsigned short* __restrict__ w2t,
              const float* __restrict__ b1, const float* __restrict__ b2,
              float* __restrict__ hout, int d_in, int n_nodes) {
  __shared__ __align__(16) unsigned short Alds[128 * HID];   // node tile (bf16), reused for H1
  __shared__ __align__(16) unsigned short Wlds[HID * HID];   // weights, transposed [N][K]

  const int tid     = threadIdx.x;
  const int row0blk = blockIdx.x * 128;

  // load A = bf16(x + agg), zero-padded; pack 2 bf16 per 32-bit LDS store
  {
    unsigned int* A32 = (unsigned int*)Alds;
    for (int idx = tid; idx < 128 * (HID / 2); idx += 256) {
      int r  = idx >> 6;              // row
      int kp = (idx & 63) * 2;        // first k of the pair
      int g  = row0blk + r;
      float v0 = 0.0f, v1 = 0.0f;
      if (g < n_nodes) {
        long o = (long)g * d_in + kp;
        if (kp     < d_in) v0 = x[o]     + agg[o];
        if (kp + 1 < d_in) v1 = x[o + 1] + agg[o + 1];
      }
      A32[idx] = (unsigned int)f32_bf16(v0) | ((unsigned int)f32_bf16(v1) << 16);
    }
  }
  // load W1 (transposed bf16) with 16-byte vector copies
  {
    const uint4* src = (const uint4*)w1t;
    uint4* dst = (uint4*)Wlds;
    for (int idx = tid; idx < (HID * HID) / 8; idx += 256) dst[idx] = src[idx];
  }
  __syncthreads();

  const int lane = tid & 31;
  const int wave = tid >> 5;
  const int half = lane >> 4;
  const int mn   = lane & 15;
  const int row0 = wave * 16;       // this wave owns rows [row0, row0+16)

  const v8f vzero = {};
  v8f acc[8];

  // ---- GEMM1: (128-row tile) x W1 ----
#pragma unroll
  for (int nt = 0; nt < 8; ++nt) acc[nt] = vzero;
  gemm128(Alds, Wlds, acc, row0, half, mn);

  // epilogue 1: +b1, relu, write bf16 H1 in place over own A rows
#pragma unroll
  for (int nt = 0; nt < 8; ++nt) {
    int n = nt * 16 + mn;
    float bias = b1[n];
#pragma unroll
    for (int r = 0; r < 8; ++r) {
      int m = r + 8 * half;
      float v = acc[nt][r] + bias;
      v = v > 0.0f ? v : 0.0f;
      Alds[(row0 + m) * HID + n] = f32_bf16(v);
    }
  }
  __syncthreads();                     // everyone done with W1
  {
    const uint4* src = (const uint4*)w2t;
    uint4* dst = (uint4*)Wlds;
    for (int idx = tid; idx < (HID * HID) / 8; idx += 256) dst[idx] = src[idx];
  }
  __syncthreads();

  // ---- GEMM2: H1 x W2 ----
#pragma unroll
  for (int nt = 0; nt < 8; ++nt) acc[nt] = vzero;
  gemm128(Alds, Wlds, acc, row0, half, mn);

  // epilogue 2: +b2, relu, store f32 h
#pragma unroll
  for (int nt = 0; nt < 8; ++nt) {
    int n = nt * 16 + mn;
    float bias = b2[n];
#pragma unroll
    for (int r = 0; r < 8; ++r) {
      int m = r + 8 * half;
      int g = row0blk + row0 + m;
      float v = acc[nt][r] + bias;
      v = v > 0.0f ? v : 0.0f;
      if (g < n_nodes) hout[(long)g * HID + n] = v;
    }
  }
}

// ---------------- BN stats: per-feature sum / sumsq ----------------
#define ROWS_PER_BLK 512
__global__ void col_reduce(const float* __restrict__ h, int n,
                           float* __restrict__ sum, float* __restrict__ sumsq) {
  int f = threadIdx.x;                       // 128 threads, one feature each
  long r0 = (long)blockIdx.x * ROWS_PER_BLK;
  long r1 = r0 + ROWS_PER_BLK; if (r1 > n) r1 = n;
  float s = 0.0f, sq = 0.0f;
  for (long r = r0; r < r1; ++r) {
    float v = h[r * HID + f];
    s += v; sq += v * v;
  }
  atomicAdd(&sum[f], s);
  atomicAdd(&sumsq[f], sq);
}

__global__ void bn_finalize(const float* __restrict__ sum, const float* __restrict__ sumsq,
                            const float* __restrict__ gamma, const float* __restrict__ beta,
                            float* __restrict__ scale, float* __restrict__ shift, int n) {
  int f = threadIdx.x;
  float inv_n = 1.0f / (float)n;
  float mean = sum[f] * inv_n;
  float var  = sumsq[f] * inv_n - mean * mean;
  float sc   = gamma[f] * rsqrtf(var + BN_EPS);
  scale[f] = sc;
  shift[f] = beta[f] - mean * sc;
}

// BN apply + feed next layer + fused segment_max into output
__global__ void bn_apply_max(const float* __restrict__ h,
                             const float* __restrict__ scale, const float* __restrict__ shift,
                             const int* __restrict__ ibatch,
                             float* __restrict__ xnext, float* __restrict__ out,
                             int n, int layer) {
  long i = (long)blockIdx.x * blockDim.x + threadIdx.x;
  long total = (long)n * HID;
  if (i >= total) return;
  int  f    = (int)(i & (HID - 1));
  long node = i >> 7;
  float v = h[i] * scale[f] + shift[f];
  xnext[i] = v;
  int g = ibatch[node];
  atomicMaxF(&out[(long)g * OUT_STRIDE + layer * HID + f], v);
}

// ---------------- orchestration ----------------
extern "C" void kernel_launch(void* const* d_in, const int* in_sizes, int n_in,
                              void* d_out, int out_size, void* d_ws, size_t ws_size,
                              hipStream_t stream) {
  const float* feat   = (const float*)d_in[0];          // 200000 x 75
  const int*   adj    = (const int*)d_in[1];            // 2 x 1.6M
  const int*   ibatch = (const int*)d_in[2];            // 200000
  // params flattened per layer: W1, b1, W2, b2, gamma, beta
  const float *W1[N_LAYERS], *B1[N_LAYERS], *W2[N_LAYERS], *B2[N_LAYERS];
  const float *GM[N_LAYERS], *BT[N_LAYERS];
  for (int l = 0; l < N_LAYERS; ++l) {
    W1[l] = (const float*)d_in[3 + 6 * l + 0];
    B1[l] = (const float*)d_in[3 + 6 * l + 1];
    W2[l] = (const float*)d_in[3 + 6 * l + 2];
    B2[l] = (const float*)d_in[3 + 6 * l + 3];
    GM[l] = (const float*)d_in[3 + 6 * l + 4];
    BT[l] = (const float*)d_in[3 + 6 * l + 5];
  }

  // workspace carve-up
  char* ws = (char*)d_ws;
  float* xbuf = (float*)ws;  ws += (size_t)N_NODES * HID * sizeof(float);
  float* agg  = (float*)ws;  ws += (size_t)N_NODES * HID * sizeof(float);
  float* hbuf = (float*)ws;  ws += (size_t)N_NODES * HID * sizeof(float);
  float* sum   = (float*)ws; ws += HID * sizeof(float);
  float* sumsq = (float*)ws; ws += HID * sizeof(float);
  float* scale = (float*)ws; ws += HID * sizeof(float);
  float* shift = (float*)ws; ws += HID * sizeof(float);
  unsigned short* wbf = (unsigned short*)ws;  // 3 layers * 2 * 128*128 bf16 (transposed)

  // pre-convert all weights to transposed padded 128x128 bf16
  for (int l = 0; l < N_LAYERS; ++l) {
    int din = (l == 0) ? IN_DIM : HID;
    cvt_w_bf16_t<<<(HID * HID + 255) / 256, 256, 0, stream>>>(W1[l], wbf + (size_t)(2 * l) * HID * HID, din);
    cvt_w_bf16_t<<<(HID * HID + 255) / 256, 256, 0, stream>>>(W2[l], wbf + (size_t)(2 * l + 1) * HID * HID, HID);
  }

  // init output for segment_max identity (-inf; empty segments keep -inf like JAX)
  {
    long n = (long)N_GRAPHS * OUT_STRIDE;
    fill_f32<<<(unsigned)((n + 255) / 256), 256, 0, stream>>>((float*)d_out, n, -INFINITY);
  }

  const float* xcur = feat;
  int d = IN_DIM;
  const int gemm_blocks = (N_NODES + 127) / 128;

  for (int l = 0; l < N_LAYERS; ++l) {
    // agg = segment_sum(x[src], dst): one block per edge, thread per feature
    long an = (long)N_NODES * d;
    fill_f32<<<(unsigned)((an + 255) / 256), 256, 0, stream>>>(agg, an, 0.0f);
    scatter_add<<<N_EDGES, d, 0, stream>>>(xcur, adj, agg, d);

    // fused MLP with bf16 WMMA
    gemm_mlp<<<gemm_blocks, 256, 0, stream>>>(
        xcur, agg,
        wbf + (size_t)(2 * l) * HID * HID, wbf + (size_t)(2 * l + 1) * HID * HID,
        B1[l], B2[l], hbuf, d, N_NODES);

    // batch-norm statistics
    fill_f32<<<1, HID, 0, stream>>>(sum, HID, 0.0f);
    fill_f32<<<1, HID, 0, stream>>>(sumsq, HID, 0.0f);
    col_reduce<<<(N_NODES + ROWS_PER_BLK - 1) / ROWS_PER_BLK, HID, 0, stream>>>(hbuf, N_NODES, sum, sumsq);
    bn_finalize<<<1, HID, 0, stream>>>(sum, sumsq, GM[l], BT[l], scale, shift, N_NODES);

    // normalize + write next-layer input + fused segment_max
    long tn = (long)N_NODES * HID;
    bn_apply_max<<<(unsigned)((tn + 255) / 256), 256, 0, stream>>>(
        hbuf, scale, shift, ibatch, xbuf, (float*)d_out, N_NODES, l);

    xcur = xbuf;
    d = HID;
  }
}
